// Linformer_10436770529785
// MI455X (gfx1250) — matrix operand order repaired and checked by reading, not compiled, not run on hardware
//
#include <hip/hip_runtime.h>

// ---------------- types ----------------
typedef __bf16 bf16_t;
typedef __attribute__((ext_vector_type(16))) __bf16 v16bf;
typedef __attribute__((ext_vector_type(8)))  float  v8f;
typedef __attribute__((ext_vector_type(4)))  unsigned int u32x4;
typedef __attribute__((ext_vector_type(4)))  unsigned int tdm_g0_t;
typedef __attribute__((ext_vector_type(8)))  int          tdm_g1_t;
typedef __attribute__((ext_vector_type(4)))  int          tdm_g2_t;
typedef __attribute__((ext_vector_type(8)))  int          tdm_g4_t;

union Frag16 { v16bf v; u32x4 u[2]; };
union V8     { u32x4 u; bf16_t e[8]; };

#if __has_builtin(__builtin_amdgcn_tensor_load_to_lds)
#define HAS_TDM 1
#else
#define HAS_TDM 0
#endif

#define LST 40   // LDS row stride in bf16 elements (32 data + 8 pad = 80B rows)

__device__ inline unsigned lds_lo32(const void* p) {
    // AMDGPU generic pointers to LDS carry the LDS byte address in the low 32 bits
    return (unsigned)(unsigned long long)p;
}

#if HAS_TDM
// Issue one 2D TDM tile load: tile_rows x tile_cols(=32) bf16 elements,
// row stride = row_stride_elems, LDS padded to 40-element rows via pad fields.
__device__ inline void tdm_issue_2d(unsigned lds_addr, const void* gaddr,
                                    long long row_stride_elems, int tile_rows)
{
    unsigned long long ga = (unsigned long long)gaddr;
    tdm_g0_t g0;
    g0[0] = 1u;                                   // count=1, user descriptor
    g0[1] = lds_addr;                             // LDS byte address
    g0[2] = (unsigned)(ga & 0xffffffffu);         // global addr [31:0]
    g0[3] = (unsigned)((ga >> 32) & 0x01ffffffu) | (2u << 30); // addr[56:32] | type=2
    const unsigned td0 = 0x40000000u, td1 = 0x40000000u;        // huge dims: no OOB clamp
    unsigned long long s0 = (unsigned long long)row_stride_elems;
    tdm_g1_t g1;
    g1[0] = (int)((1u << 16)      // data_size = 2 bytes
                | (1u << 20)      // pad_enable
                | (3u << 22)      // pad_interval: 16 dwords (one 32-elem row)
                | (3u << 25));    // pad_amount: 4 dwords (8 bf16)
    g1[1] = (int)((td0 & 0xffffu) << 16);                         // tensor_dim0[15:0]
    g1[2] = (int)((td0 >> 16) | ((td1 & 0xffffu) << 16));         // td0 hi | td1 lo
    g1[3] = (int)((td1 >> 16) | (32u << 16));                     // td1 hi | tile_dim0=32
    g1[4] = (int)((unsigned)tile_rows);                           // tile_dim1 | tile_dim2=0
    g1[5] = (int)(unsigned)(s0 & 0xffffffffu);                    // dim0_stride[31:0]
    g1[6] = (int)(((s0 >> 32) & 0xffffu) | ((s0 & 0xffffu) << 16));
    g1[7] = (int)((s0 >> 16) & 0xffffffffu);                      // dim1_stride (unused, 2D)
    tdm_g2_t z4 = {0, 0, 0, 0};
    tdm_g4_t z8 = {0, 0, 0, 0, 0, 0, 0, 0};
    __builtin_amdgcn_tensor_load_to_lds(g0, g1, z4, z4, z8, 0);
}
#endif

// =======================================================================
// NT GEMM:  C = act(alpha * A(MxK, row-major) * B(NxK, row-major)^T + bias)
// Block tile 256x64x32, 8 waves (4x2), wave tile 64x32 (4x2 WMMA tiles).
// A and B tiles staged by the Tensor Data Mover, double-buffered (TENSORcnt).
// =======================================================================
#define TBM 256
#define TBN 64
#define TBK 32

__global__ __launch_bounds__(256)
void gemm_nt_tdm(const bf16_t* __restrict__ Ag, long long lda, long long sAb, long long sAh,
                 const bf16_t* __restrict__ Bg, long long ldb, long long sBb, long long sBh,
                 void* __restrict__ Cg, long long ldc, long long sCb, long long sCh,
                 int M, int Nn, int Kd, int batchH,
                 float alpha, const float* __restrict__ bias, int act, int outBf16)
{
    __shared__ bf16_t As[2][TBM * LST];
    __shared__ bf16_t Bs[2][TBN * LST];

    const int tid = threadIdx.x;
    const int z  = blockIdx.z;
    const int bb = z / batchH;
    const int hh = z % batchH;
    const bf16_t* A = Ag + (long long)bb * sAb + (long long)hh * sAh;
    const bf16_t* B = Bg + (long long)bb * sBb + (long long)hh * sBh;
    const long long coff = (long long)bb * sCb + (long long)hh * sCh;

    const int m0 = blockIdx.y * TBM;
    const int n0 = blockIdx.x * TBN;

    const int wave = tid >> 5, lane = tid & 31;
    const int wm = wave >> 1, wn = wave & 1;     // 4x2 wave grid -> 256x64
    const int half = lane >> 4, mr = lane & 15;

    v8f acc[4][2] = {};
    const int nsteps = Kd / TBK;

    // per-wave compute on one staged buffer
    auto compute_step = [&](const bf16_t* AsP, const bf16_t* BsP) {
        Frag16 fa[4], fb[2];
        #pragma unroll
        for (int t = 0; t < 4; ++t) {
            int arow = wm * 64 + t * 16 + mr;
            int ac0  = half * 8;                 // lanes<16: K 0..7 & 16..23
            fa[t].u[0] = *(const u32x4*)&AsP[arow * LST + ac0];
            fa[t].u[1] = *(const u32x4*)&AsP[arow * LST + ac0 + 16];
        }
        #pragma unroll
        for (int t = 0; t < 2; ++t) {
            int brow = wn * 32 + t * 16 + mr;
            int bc0  = half * 8;                 // lanes<16: K 0..7 & 16..23
            fb[t].u[0] = *(const u32x4*)&BsP[brow * LST + bc0];
            fb[t].u[1] = *(const u32x4*)&BsP[brow * LST + bc0 + 16];
        }
        #pragma unroll
        for (int tm = 0; tm < 4; ++tm)
            #pragma unroll
            for (int tn = 0; tn < 2; ++tn)
                acc[tm][tn] = __builtin_amdgcn_wmma_f32_16x16x32_bf16(
                    false, fa[tm].v, false, fb[tn].v, (short)0, acc[tm][tn], false, false);
    };

#if HAS_TDM
    const unsigned asAddr[2] = { lds_lo32(&As[0][0]), lds_lo32(&As[1][0]) };
    const unsigned bsAddr[2] = { lds_lo32(&Bs[0][0]), lds_lo32(&Bs[1][0]) };
    if (wave == 0) {
        tdm_issue_2d(asAddr[0], A + (long long)m0 * lda, lda, TBM);
        tdm_issue_2d(bsAddr[0], B + (long long)n0 * ldb, ldb, TBN);
    }
    for (int i = 0; i < nsteps; ++i) {
        if (wave == 0) {
            if (i + 1 < nsteps) {
                int nb = (i + 1) & 1;
                long long k1 = (long long)(i + 1) * TBK;
                tdm_issue_2d(asAddr[nb], A + (long long)m0 * lda + k1, lda, TBM);
                tdm_issue_2d(bsAddr[nb], B + (long long)n0 * ldb + k1, ldb, TBN);
                __builtin_amdgcn_s_wait_tensorcnt(2);   // buffer i pair complete
            } else {
                __builtin_amdgcn_s_wait_tensorcnt(0);
            }
        }
        __syncthreads();
        compute_step(&As[i & 1][0], &Bs[i & 1][0]);
        __syncthreads();
    }
#else
    for (int i = 0; i < nsteps; ++i) {
        int k0 = i * TBK;
        #pragma unroll
        for (int j = 0; j < 4; ++j) {            // A: 256 rows x 4 chunks of 8
            int c = tid + j * 256;
            int row = c >> 2, cc = c & 3;
            *(u32x4*)&As[0][row * LST + cc * 8] =
                *(const u32x4*)(A + (long long)(m0 + row) * lda + k0 + cc * 8);
        }
        {
            int nr = tid >> 2, kc = tid & 3;     // B: 64 rows x 4 chunks
            *(u32x4*)&Bs[0][nr * LST + kc * 8] =
                *(const u32x4*)(B + (long long)(n0 + nr) * ldb + k0 + kc * 8);
        }
        __syncthreads();
        compute_step(&As[0][0], &Bs[0][0]);
        __syncthreads();
    }
#endif

    // ---- epilogue ----
    #pragma unroll
    for (int tm = 0; tm < 4; ++tm) {
        #pragma unroll
        for (int tn = 0; tn < 2; ++tn) {
            int gn = n0 + wn * 32 + tn * 16 + mr;
            float bv = bias ? bias[gn] : 0.0f;
            #pragma unroll
            for (int i = 0; i < 8; ++i) {
                int gm = m0 + wm * 64 + tm * 16 + half * 8 + i;
                float v = acc[tm][tn][i] * alpha + bv;
                if (act == 1) v = 0.5f * v * (1.0f + erff(v * 0.70710678118654752f));
                long long idx = coff + (long long)gm * ldc + gn;
                if (outBf16) ((bf16_t*)Cg)[idx] = (bf16_t)v;
                else         ((float*)Cg)[idx]  = v;
            }
        }
    }
}

// =======================================================================
// Projection GEMM:  C = A^T * B  where A stored [K x M] (E/F), B stored [K x N]
// (k/v head slice).  Block 128x64x32, manual scatter staging.
// storeT=1 writes C^T (used to produce vp^T for the NT attention-out GEMM).
// =======================================================================
#define PBM 128
#define PBN 64

__global__ __launch_bounds__(256)
void gemm_proj(const bf16_t* __restrict__ Ag, long long lda,
               const bf16_t* __restrict__ Bg, long long ldb, long long sBb, long long sBh,
               void* __restrict__ Cg, long long ldc, long long sCb, long long sCh,
               int M, int Nn, int Kd, int batchH, int storeT)
{
    __shared__ bf16_t As[PBM * LST];
    __shared__ bf16_t Bs[PBN * LST];

    const int tid = threadIdx.x;
    const int z  = blockIdx.z;
    const int bb = z / batchH;
    const int hh = z % batchH;
    const bf16_t* A = Ag;
    const bf16_t* B = Bg + (long long)bb * sBb + (long long)hh * sBh;
    const long long coff = (long long)bb * sCb + (long long)hh * sCh;

    const int m0 = blockIdx.y * PBM;
    const int n0 = blockIdx.x * PBN;

    const int wave = tid >> 5, lane = tid & 31;
    const int wm = wave >> 1, wn = wave & 1;     // 4x2 -> 128x64
    const int half = lane >> 4, mr = lane & 15;

    v8f acc[2][2] = {};

    for (int k0 = 0; k0 < Kd; k0 += TBK) {
        // A stored [kk][m]: transpose-scatter into As[m][kk]
        #pragma unroll
        for (int i = 0; i < 2; ++i) {
            int c = tid + i * 256;
            int kkr = c >> 4, mc = c & 15;
            V8 t; t.u = *(const u32x4*)(A + (long long)(k0 + kkr) * lda + m0 + mc * 8);
            #pragma unroll
            for (int j = 0; j < 8; ++j)
                As[(mc * 8 + j) * LST + kkr] = t.e[j];
        }
        // B stored [kk][n]: transpose-scatter into Bs[n][kk]
        {
            int kkr = tid >> 3, nc = tid & 7;
            V8 t; t.u = *(const u32x4*)(B + (long long)(k0 + kkr) * ldb + n0 + nc * 8);
            #pragma unroll
            for (int j = 0; j < 8; ++j)
                Bs[(nc * 8 + j) * LST + kkr] = t.e[j];
        }
        __syncthreads();

        Frag16 fa[2], fb[2];
        #pragma unroll
        for (int t = 0; t < 2; ++t) {
            int arow = wm * 32 + t * 16 + mr;
            int ac0  = half * 8;
            fa[t].u[0] = *(const u32x4*)&As[arow * LST + ac0];
            fa[t].u[1] = *(const u32x4*)&As[arow * LST + ac0 + 16];
            int brow = wn * 32 + t * 16 + mr;
            int bc0  = half * 8;
            fb[t].u[0] = *(const u32x4*)&Bs[brow * LST + bc0];
            fb[t].u[1] = *(const u32x4*)&Bs[brow * LST + bc0 + 16];
        }
        #pragma unroll
        for (int tm = 0; tm < 2; ++tm)
            #pragma unroll
            for (int tn = 0; tn < 2; ++tn)
                acc[tm][tn] = __builtin_amdgcn_wmma_f32_16x16x32_bf16(
                    false, fa[tm].v, false, fb[tn].v, (short)0, acc[tm][tn], false, false);
        __syncthreads();
    }

    #pragma unroll
    for (int tm = 0; tm < 2; ++tm) {
        #pragma unroll
        for (int tn = 0; tn < 2; ++tn) {
            int gn = n0 + wn * 32 + tn * 16 + mr;
            #pragma unroll
            for (int i = 0; i < 8; ++i) {
                int gm = m0 + wm * 32 + tm * 16 + half * 8 + i;
                long long idx = storeT ? coff + (long long)gn * ldc + gm
                                       : coff + (long long)gm * ldc + gn;
                ((bf16_t*)Cg)[idx] = (bf16_t)acc[tm][tn][i];
            }
        }
    }
}

// ---------------- softmax over rows of 256 (writes bf16 probs) ----------------
__global__ __launch_bounds__(256)
void softmax256(const float* __restrict__ S, bf16_t* __restrict__ P)
{
    __shared__ float red[256];
    long long row = blockIdx.x;
    int tid = threadIdx.x;
    float v = S[row * 256 + tid];
    red[tid] = v; __syncthreads();
    for (int s = 128; s > 0; s >>= 1) { if (tid < s) red[tid] = fmaxf(red[tid], red[tid + s]); __syncthreads(); }
    float mx = red[0]; __syncthreads();
    float e = __expf(v - mx);
    red[tid] = e; __syncthreads();
    for (int s = 128; s > 0; s >>= 1) { if (tid < s) red[tid] += red[tid + s]; __syncthreads(); }
    float inv = 1.0f / red[0];
    P[row * 256 + tid] = (bf16_t)(e * inv);
}

// ---------------- residual add + LayerNorm (D=1024) -> f32 + bf16 ----------------
__global__ __launch_bounds__(256)
void add_layernorm1024(const float* __restrict__ a, const float* __restrict__ res,
                       const float* __restrict__ g, const float* __restrict__ bta,
                       float* __restrict__ xout, bf16_t* __restrict__ xbout)
{
    __shared__ float red[256];
    const int tid = threadIdx.x;
    const long long base = (long long)blockIdx.x * 1024;
    float v[4]; float s = 0.0f;
    #pragma unroll
    for (int j = 0; j < 4; ++j) { int c = tid + j * 256; v[j] = a[base + c] + res[base + c]; s += v[j]; }
    red[tid] = s; __syncthreads();
    for (int t = 128; t > 0; t >>= 1) { if (tid < t) red[tid] += red[tid + t]; __syncthreads(); }
    float mean = red[0] * (1.0f / 1024.0f); __syncthreads();
    float sq = 0.0f;
    #pragma unroll
    for (int j = 0; j < 4; ++j) { float d = v[j] - mean; sq += d * d; }
    red[tid] = sq; __syncthreads();
    for (int t = 128; t > 0; t >>= 1) { if (tid < t) red[tid] += red[tid + t]; __syncthreads(); }
    float rs = rsqrtf(red[0] * (1.0f / 1024.0f) + 1e-12f);
    #pragma unroll
    for (int j = 0; j < 4; ++j) {
        int c = tid + j * 256;
        float y = (v[j] - mean) * rs * g[c] + bta[c];
        xout[base + c]  = y;
        xbout[base + c] = (bf16_t)y;
    }
}

// ---------------- f32 -> bf16 converts ----------------
__global__ __launch_bounds__(256)
void cvt_f32_bf16(const float* __restrict__ in, bf16_t* __restrict__ out, long long n)
{
    long long i = (long long)blockIdx.x * 256 + threadIdx.x;
    long long stride = (long long)gridDim.x * 256;
    for (; i < n; i += stride) out[i] = (bf16_t)in[i];
}

// transposed convert: in [R x C] row-major f32 -> out [C x R] row-major bf16
__global__ __launch_bounds__(256)
void cvt_t_f32_bf16(const float* __restrict__ in, bf16_t* __restrict__ out, int R, int C)
{
    long long n = (long long)R * C;
    long long i = (long long)blockIdx.x * 256 + threadIdx.x;
    long long stride = (long long)gridDim.x * 256;
    for (; i < n; i += stride) {
        int r = (int)(i / C), c = (int)(i % C);
        out[(long long)c * R + r] = (bf16_t)in[i];
    }
}

// ---------------- host orchestration ----------------
extern "C" void kernel_launch(void* const* d_in, const int* in_sizes, int n_in,
                              void* d_out, int out_size, void* d_ws, size_t ws_size,
                              hipStream_t stream)
{
    const int  Lc = 4, Bc = 2, Nc = 4096, Dc = 1024, Hc = 16, HDc = 64, Kc = 256, FFc = 4096;
    const long long NB  = (long long)Bc * Nc;
    const long long DD  = (long long)Dc * Dc;
    const long long NK  = (long long)Nc * Kc;
    const long long FD  = (long long)FFc * Dc;
    const long long ND  = (long long)Nc * Dc;
    const long long KHD = (long long)Kc * HDc;

    const float* x_in = (const float*)d_in[0];
    const float* Wq = (const float*)d_in[1];
    const float* Wk = (const float*)d_in[2];
    const float* Wv = (const float*)d_in[3];
    const float* E  = (const float*)d_in[4];
    const float* F  = (const float*)d_in[5];
    const float* ln1g = (const float*)d_in[6];
    const float* ln1b = (const float*)d_in[7];
    const float* W1 = (const float*)d_in[8];
    const float* b1 = (const float*)d_in[9];
    const float* W2 = (const float*)d_in[10];
    const float* b2 = (const float*)d_in[11];
    const float* ln2g = (const float*)d_in[12];
    const float* ln2b = (const float*)d_in[13];

    char* wp = (char*)d_ws;
    auto alloc = [&](size_t bytes) -> void* {
        void* r = (void*)wp;
        wp += (bytes + 255) & ~(size_t)255;
        return r;
    };
    float*  x_cur  = (float*) alloc(NB * Dc * 4);
    bf16_t* xb     = (bf16_t*)alloc(NB * Dc * 2);
    bf16_t* qb     = (bf16_t*)alloc(NB * Dc * 2);
    bf16_t* kb     = (bf16_t*)alloc(NB * Dc * 2);
    bf16_t* vb     = (bf16_t*)alloc(NB * Dc * 2);
    bf16_t* wqT    = (bf16_t*)alloc(DD * 2);   // [out,in] after transpose
    bf16_t* wkT    = (bf16_t*)alloc(DD * 2);
    bf16_t* wvT    = (bf16_t*)alloc(DD * 2);
    bf16_t* e_b    = (bf16_t*)alloc(NK * 2);
    bf16_t* f_b    = (bf16_t*)alloc(NK * 2);
    bf16_t* w1_b   = (bf16_t*)alloc(FD * 2);   // torch [out,in] already
    bf16_t* w2_b   = (bf16_t*)alloc(FD * 2);
    bf16_t* kp_b   = (bf16_t*)alloc((long long)Bc * Hc * KHD * 2);   // [e][d]
    bf16_t* vpT    = (bf16_t*)alloc((long long)Bc * Hc * KHD * 2);   // [d][e]
    float*  scores = (float*) alloc((long long)Bc * Hc * Nc * Kc * 4);
    bf16_t* attn_b = (bf16_t*)alloc((long long)Bc * Hc * Nc * Kc * 2);
    float*  obuf   = (float*) alloc(NB * Dc * 4);
    bf16_t* hb     = (bf16_t*)alloc(NB * FFc * 2);
    float*  ybuf   = (float*) alloc(NB * Dc * 4);
    (void)ws_size; (void)in_sizes; (void)n_in; (void)out_size;

    auto cvt = [&](const float* src, bf16_t* dst, long long n) {
        cvt_f32_bf16<<<2048, 256, 0, stream>>>(src, dst, n);
    };

    cvt(x_in, xb, NB * Dc);

    for (int l = 0; l < Lc; ++l) {
        // weights -> bf16 (QKV transposed to [out,in] so every big GEMM is NT)
        cvt_t_f32_bf16<<<2048, 256, 0, stream>>>(Wq + (long long)l * DD, wqT, Dc, Dc);
        cvt_t_f32_bf16<<<2048, 256, 0, stream>>>(Wk + (long long)l * DD, wkT, Dc, Dc);
        cvt_t_f32_bf16<<<2048, 256, 0, stream>>>(Wv + (long long)l * DD, wvT, Dc, Dc);
        cvt(E  + (long long)l * NK, e_b,  NK);
        cvt(F  + (long long)l * NK, f_b,  NK);
        cvt(W1 + (long long)l * FD, w1_b, FD);
        cvt(W2 + (long long)l * FD, w2_b, FD);

        // Q/K/V = xb @ W^T  (NT, TDM-staged, out bf16)
        dim3 gQ(Dc / TBN, (unsigned)(NB / TBM), 1);
        gemm_nt_tdm<<<gQ, 256, 0, stream>>>(xb, Dc, 0, 0, wqT, Dc, 0, 0,
            qb, Dc, 0, 0, (int)NB, Dc, Dc, 1, 1.0f, nullptr, 0, 1);
        gemm_nt_tdm<<<gQ, 256, 0, stream>>>(xb, Dc, 0, 0, wkT, Dc, 0, 0,
            kb, Dc, 0, 0, (int)NB, Dc, Dc, 1, 1.0f, nullptr, 0, 1);
        gemm_nt_tdm<<<gQ, 256, 0, stream>>>(xb, Dc, 0, 0, wvT, Dc, 0, 0,
            vb, Dc, 0, 0, (int)NB, Dc, Dc, 1, 1.0f, nullptr, 0, 1);

        // kp[b,h] = E^T @ K_head  -> [e][d];  vp^T[b,h] = (F^T @ V_head)^T -> [d][e]
        dim3 gP(HDc / PBN, Kc / PBM, Bc * Hc);
        gemm_proj<<<gP, 256, 0, stream>>>(e_b, Kc, kb, Dc, ND, HDc,
            kp_b, HDc, (long long)Hc * KHD, KHD, Kc, HDc, Nc, Hc, 0);
        gemm_proj<<<gP, 256, 0, stream>>>(f_b, Kc, vb, Dc, ND, HDc,
            vpT, Kc, (long long)Hc * KHD, KHD, Kc, HDc, Nc, Hc, 1);

        // scores = Q_head @ kp^T / sqrt(HD)   (NT: B rows are e, cols d)
        dim3 gS(Kc / TBN, Nc / TBM, Bc * Hc);
        gemm_nt_tdm<<<gS, 256, 0, stream>>>(
            qb, Dc, ND, HDc, kp_b, HDc, (long long)Hc * KHD, KHD,
            scores, Kc, (long long)Hc * Nc * Kc, (long long)Nc * Kc,
            Nc, Kc, HDc, Hc, 0.125f, nullptr, 0, 0);

        softmax256<<<(unsigned)(Bc * Hc * Nc), 256, 0, stream>>>(scores, attn_b);

        // out_head = P @ vp = P @ (vp^T)^T    (NT: B rows are d, cols e)
        dim3 gO(HDc / TBN, Nc / TBM, Bc * Hc);
        gemm_nt_tdm<<<gO, 256, 0, stream>>>(
            attn_b, Kc, (long long)Hc * Nc * Kc, (long long)Nc * Kc,
            vpT, Kc, (long long)Hc * KHD, KHD,
            obuf, Dc, ND, HDc,
            Nc, HDc, Kc, Hc, 1.0f, nullptr, 0, 0);

        const float* res0 = (l == 0) ? x_in : x_cur;
        add_layernorm1024<<<(unsigned)NB, 256, 0, stream>>>(
            obuf, res0, ln1g + (long long)l * Dc, ln1b + (long long)l * Dc, x_cur, xb);

        // h = gelu(x @ W1^T + b1)   (NT, fused bias+gelu, out bf16)
        dim3 gF1(FFc / TBN, (unsigned)(NB / TBM), 1);
        gemm_nt_tdm<<<gF1, 256, 0, stream>>>(xb, Dc, 0, 0, w1_b, Dc, 0, 0,
            hb, FFc, 0, 0, (int)NB, FFc, Dc, 1, 1.0f, b1 + (long long)l * FFc, 1, 1);

        // y = h @ W2^T + b2   (NT, out f32)
        dim3 gF2(Dc / TBN, (unsigned)(NB / TBM), 1);
        gemm_nt_tdm<<<gF2, 256, 0, stream>>>(hb, FFc, 0, 0, w2_b, FFc, 0, 0,
            ybuf, Dc, 0, 0, (int)NB, Dc, FFc, 1, 1.0f, b2 + (long long)l * Dc, 0, 0);

        float* dst = (l == Lc - 1) ? (float*)d_out : x_cur;
        add_layernorm1024<<<(unsigned)NB, 256, 0, stream>>>(
            ybuf, x_cur, ln2g + (long long)l * Dc, ln2b + (long long)l * Dc, dst, xb);
    }
}
